// MILRNN_8143257993451
// MI455X (gfx1250) — matrix operand order repaired and checked by reading, not compiled, not run on hardware
//
#include <hip/hip_runtime.h>

// ---------------------------------------------------------------------------
// Problem constants (from reference): x[64,512,1024], W1[512,1024], W2[512,512]
// ---------------------------------------------------------------------------
constexpr int EMB = 512;   // embed dim D
constexpr int NIN = 1024;  // input feature dim N
constexpr int NB  = 64;    // batch B
constexpr int NL  = 512;   // sequence length L
constexpr int NBLK_RNN = EMB / 32;  // 16 persistent blocks for the recurrence

typedef __attribute__((ext_vector_type(16))) __bf16 v16bf;
typedef __attribute__((ext_vector_type(8)))  __bf16 v8bf;
typedef __attribute__((ext_vector_type(8)))  float  v8f;
typedef __attribute__((ext_vector_type(4)))  float  v4f;

// ---------------------------------------------------------------------------
// Kernel 0: one-shot fp32 -> bf16 weight conversion (W1, W2).
// ---------------------------------------------------------------------------
__global__ __launch_bounds__(256) void cvt_bf16_kernel(const float* __restrict__ src,
                                                       __bf16* __restrict__ dst,
                                                       int n8)
{
    const int i = blockIdx.x * 256 + threadIdx.x;
    if (i < n8) {
        const float* p = src + (size_t)i * 8;
        v4f f0 = *(const v4f*)p;
        v4f f1 = *(const v4f*)(p + 4);
        v8bf o;
#pragma unroll
        for (int e = 0; e < 4; ++e) {
            o[e]     = (__bf16)f0[e];
            o[4 + e] = (__bf16)f1[e];
        }
        *(v8bf*)(dst + (size_t)i * 8) = o;
    }
}

// ---------------------------------------------------------------------------
// Kernel 1: fc1  —  xp[l][b][d] = bf16( x[b,l,:] . W1[d,:] + b1[d] )
// One block per M-tile (16 rows of flattened [B*L, N]); 8 waves; each wave
// owns 4 D-tiles. Software-pipelined K loop: fragments for chunk k+1 are
// issued before the 4 WMMAs of chunk k so global latency overlaps WMMA.
// x is read exactly once from HBM; W1bf (1 MB) lives in L2.
// ---------------------------------------------------------------------------
__device__ __forceinline__ v16bf fc1_load_a(const float* xRow, int k0, int hi)
{
    const float* ap = xRow + k0 + hi * 8;
    v4f a0 = *(const v4f*)(ap);
    v4f a1 = *(const v4f*)(ap + 4);
    v4f a2 = *(const v4f*)(ap + 16);
    v4f a3 = *(const v4f*)(ap + 20);
    v16bf a;
#pragma unroll
    for (int i = 0; i < 4; ++i) {
        a[i]      = (__bf16)a0[i];
        a[4 + i]  = (__bf16)a1[i];
        a[8 + i]  = (__bf16)a2[i];
        a[12 + i] = (__bf16)a3[i];
    }
    return a;
}

__device__ __forceinline__ v16bf fc1_load_b(const __bf16* wRow, int k0, int hi)
{
    const __bf16* bp = wRow + k0 + hi * 8;
    v8bf w0 = *(const v8bf*)(bp);
    v8bf w1 = *(const v8bf*)(bp + 16);
    v16bf b;
#pragma unroll
    for (int e = 0; e < 8; ++e) { b[e] = w0[e]; b[8 + e] = w1[e]; }
    return b;
}

__global__ __launch_bounds__(256) void fc1_kernel(const float* __restrict__ x,
                                                  const __bf16* __restrict__ W1bf,
                                                  const float* __restrict__ b1,
                                                  __bf16* __restrict__ xp)
{
    const int tid  = threadIdx.x;
    const int wid  = tid >> 5;
    const int lane = tid & 31;
    const int hi   = (lane >> 4) & 1;
    const int r    = lane & 15;
    const int mBase = blockIdx.x * 16;

    v8f acc[4] = {};
    const float* xRow = x + (size_t)(mBase + r) * NIN;
    const __bf16* wRow[4];
#pragma unroll
    for (int i = 0; i < 4; ++i)
        wRow[i] = W1bf + (size_t)((wid + i * 8) * 16 + r) * NIN;

    // Prologue: fragments for chunk 0.
    v16bf a = fc1_load_a(xRow, 0, hi);
    v16bf bfr[4];
#pragma unroll
    for (int i = 0; i < 4; ++i) bfr[i] = fc1_load_b(wRow[i], 0, hi);

    // Pipelined main loop: issue chunk k+32 loads, then WMMA chunk k.
    for (int k0 = 0; k0 < NIN - 32; k0 += 32) {
        const int kn = k0 + 32;
        __builtin_prefetch(xRow + kn + 32, 0, 3);  // chunk k+64 into near cache
        v16bf aN = fc1_load_a(xRow, kn, hi);
        v16bf bN[4];
#pragma unroll
        for (int i = 0; i < 4; ++i) bN[i] = fc1_load_b(wRow[i], kn, hi);
#pragma unroll
        for (int i = 0; i < 4; ++i)
            acc[i] = __builtin_amdgcn_wmma_f32_16x16x32_bf16(
                false, a, false, bfr[i], (short)0, acc[i], false, false);
        a = aN;
#pragma unroll
        for (int i = 0; i < 4; ++i) bfr[i] = bN[i];
    }
    // Epilogue chunk.
#pragma unroll
    for (int i = 0; i < 4; ++i)
        acc[i] = __builtin_amdgcn_wmma_f32_16x16x32_bf16(
            false, a, false, bfr[i], (short)0, acc[i], false, false);

    // + b1, store bf16 time-major [L, B, D] for the recurrence.
    // C/D layout: element j -> row M=j+hi*8, col N=lane&15.
#pragma unroll
    for (int i = 0; i < 4; ++i) {
        const int d = (wid + i * 8) * 16 + r;
        const float bias = b1[d];
#pragma unroll
        for (int j = 0; j < 8; ++j) {
            const int m  = mBase + j + hi * 8;  // flattened b*L + l
            const int bb = m >> 9;
            const int l  = m & (NL - 1);
            xp[(size_t)l * (NB * EMB) + (size_t)bb * EMB + d] =
                (__bf16)(acc[i][j] + bias);
        }
    }
}

// ---------------------------------------------------------------------------
// Kernel 2: recurrence — persistent kernel, 16 blocks x 256 threads.
// Block g owns output columns [g*32, g*32+32); its bf16 W2 slab (32x512) is
// staged once into LDS. Per step, the whole 64x512 bf16 state is pulled into
// LDS with global_load_async_to_lds_b128 (ASYNCcnt), then each of the 8
// waves computes one 16x16 tile of the new state (K=512 -> 16 WMMAs) with
// A fragments from LDS (ds_load_b128) and B fragments from the W2 slab.
// State is double-buffered bf16 in global; a spin/atomic grid barrier
// separates the 512 steps. Step 0 uses state==0 implicitly (no GEMM), so the
// workspace needs no zero-init.
// ---------------------------------------------------------------------------
__global__ __launch_bounds__(256) void rnn_kernel(const __bf16* __restrict__ W2bf,
                                                  const float* __restrict__ b2,
                                                  const __bf16* __restrict__ xp,
                                                  __bf16* __restrict__ s0,
                                                  __bf16* __restrict__ s1,
                                                  float* __restrict__ stateF,
                                                  unsigned* __restrict__ counter)
{
    constexpr int WSTRIDE = 536;  // bf16 elems; 1072 B row stride (16B aligned)
    constexpr int SSTRIDE = 528;  // bf16 elems; 1056 B row stride (16B aligned)
    __shared__ __bf16 ldsW2[32 * WSTRIDE];
    __shared__ __bf16 ldsS[64 * SSTRIDE];

    const int tid  = threadIdx.x;
    const int wid  = tid >> 5;
    const int lane = tid & 31;
    const int hi   = (lane >> 4) & 1;
    const int r    = lane & 15;
    const int bRow = wid & 3;            // batch tile 0..3
    const int dLoc = (wid >> 2) * 16;    // 0 or 16 within this block's slab
    const int dBlk = blockIdx.x * 32;

    // Stage this block's 32 bf16 rows of W2 into LDS (vector copies).
    for (int idx = tid; idx < 32 * (EMB / 8); idx += 256) {
        const int row = idx >> 6;
        const int c8  = (idx & 63) * 8;
        *(v8bf*)&ldsW2[row * WSTRIDE + c8] =
            *(const v8bf*)&W2bf[(size_t)(dBlk + row) * EMB + c8];
    }
    __syncthreads();

    const int d      = dBlk + dLoc + r;  // this lane's output column
    const float bias = b2[d];
    __bf16* bufs[2]  = { s0, s1 };

    for (int t = 0; t < NL; ++t) {
        v8f acc = {};
        if (t > 0) {
            const __bf16* cur = bufs[(t - 1) & 1];
            // Async-copy the full state (64x512 bf16 = 4096 b128 segments)
            // into LDS; tracked by ASYNCcnt.
            for (int seg = tid; seg < 64 * (EMB / 8); seg += 256) {
                const int row = seg >> 6;
                const int cE  = (seg & 63) * 8;
                unsigned ldsOff = (unsigned)(uintptr_t)(const void*)
                                  &ldsS[row * SSTRIDE + cE];
                const __bf16* gp = cur + (size_t)row * EMB + cE;
                asm volatile("global_load_async_to_lds_b128 %0, %1, off"
                             :: "v"(ldsOff), "v"(gp) : "memory");
            }
            asm volatile("s_wait_asynccnt 0x0" ::: "memory");
            __syncthreads();

            const __bf16* aRowp = &ldsS[(bRow * 16 + r) * SSTRIDE];
#pragma unroll 4
            for (int k0 = 0; k0 < EMB; k0 += 32) {
                const __bf16* ap = aRowp + k0 + hi * 8;
                v8bf a0 = *(const v8bf*)(ap);        // K hi*8 .. +7
                v8bf a1 = *(const v8bf*)(ap + 16);   // K 16+hi*8 .. +7
                v16bf a;
#pragma unroll
                for (int e = 0; e < 8; ++e) { a[e] = a0[e]; a[8 + e] = a1[e]; }

                const __bf16* bp = &ldsW2[(dLoc + r) * WSTRIDE + k0 + hi * 8];
                v8bf w0 = *(const v8bf*)(bp);
                v8bf w1 = *(const v8bf*)(bp + 16);
                v16bf bfrag;
#pragma unroll
                for (int e = 0; e < 8; ++e) { bfrag[e] = w0[e]; bfrag[8 + e] = w1[e]; }

                acc = __builtin_amdgcn_wmma_f32_16x16x32_bf16(
                    false, a, false, bfrag, (short)0, acc, false, false);
            }
        }

        // Epilogue: relu(acc + b2 + xp[t]) -> next state (bf16).
        __bf16* nxt       = bufs[t & 1];
        const __bf16* xpt = xp + (size_t)t * (NB * EMB);
#pragma unroll
        for (int j = 0; j < 8; ++j) {
            const int row = bRow * 16 + j + hi * 8;
            float v = acc[j] + bias + (float)xpt[(size_t)row * EMB + d];
            v = v > 0.0f ? v : 0.0f;
            nxt[(size_t)row * EMB + d] = (__bf16)v;
            if (t == NL - 1) stateF[(size_t)row * EMB + d] = v;
        }

        // Device-wide barrier between steps (16 co-resident blocks).
        __threadfence();
        __syncthreads();
        if (tid == 0) {
            atomicAdd(counter, 1u);
            const unsigned target = (unsigned)NBLK_RNN * (unsigned)(t + 1);
            while (__atomic_load_n(counter, __ATOMIC_RELAXED) < target) {
                __builtin_amdgcn_s_sleep(2);
            }
        }
        __syncthreads();
        __threadfence();
    }
}

// ---------------------------------------------------------------------------
// Kernel 3: fc3 — out[b,c] = stateF[b,:] . W3[c,:] + b3[c]   (128 dots of 512)
// ---------------------------------------------------------------------------
__global__ __launch_bounds__(128) void fc3_kernel(const float* __restrict__ stateF,
                                                  const float* __restrict__ W3,
                                                  const float* __restrict__ b3,
                                                  float* __restrict__ out)
{
    const int tid = threadIdx.x;
    const int b = tid >> 1;
    const int c = tid & 1;
    const float* s = stateF + (size_t)b * EMB;
    const float* w = W3 + (size_t)c * EMB;
    float acc = 0.0f;
    for (int i = 0; i < EMB; i += 4) {
        v4f sv = *(const v4f*)(s + i);
        v4f wv = *(const v4f*)(w + i);
        acc += sv[0] * wv[0] + sv[1] * wv[1] + sv[2] * wv[2] + sv[3] * wv[3];
    }
    out[b * 2 + c] = acc + b3[c];
}

// ---------------------------------------------------------------------------
// Host entry
// ---------------------------------------------------------------------------
extern "C" void kernel_launch(void* const* d_in, const int* in_sizes, int n_in,
                              void* d_out, int out_size, void* d_ws, size_t ws_size,
                              hipStream_t stream)
{
    (void)in_sizes; (void)n_in; (void)out_size; (void)ws_size;

    const float* x  = (const float*)d_in[0];
    const float* W1 = (const float*)d_in[1];
    const float* b1 = (const float*)d_in[2];
    const float* W2 = (const float*)d_in[3];
    const float* b2 = (const float*)d_in[4];
    const float* W3 = (const float*)d_in[5];
    const float* b3 = (const float*)d_in[6];
    float* out = (float*)d_out;

    char* ws = (char*)d_ws;
    size_t off = 0;
    __bf16* xp = (__bf16*)(ws + off);     off += (size_t)NL * NB * EMB * 2;  // 32 MB
    __bf16* s0 = (__bf16*)(ws + off);     off += (size_t)NB * EMB * 2;
    __bf16* s1 = (__bf16*)(ws + off);     off += (size_t)NB * EMB * 2;
    float*  stateF = (float*)(ws + off);  off += (size_t)NB * EMB * 4;
    unsigned* counter = (unsigned*)(ws + off); off += 256;
    __bf16* W1bf = (__bf16*)(ws + off);   off += (size_t)EMB * NIN * 2;      // 1 MB
    __bf16* W2bf = (__bf16*)(ws + off);   off += (size_t)EMB * EMB * 2;      // 0.5 MB

    // Grid-barrier counter must start at 0 every launch (capturable memset).
    hipMemsetAsync(counter, 0, 64, stream);

    // One-shot weight conversion to bf16.
    cvt_bf16_kernel<<<(EMB * NIN / 8 + 255) / 256, 256, 0, stream>>>(W1, W1bf, EMB * NIN / 8);
    cvt_bf16_kernel<<<(EMB * EMB / 8 + 255) / 256, 256, 0, stream>>>(W2, W2bf, EMB * EMB / 8);

    fc1_kernel<<<(NB * NL) / 16, 256, 0, stream>>>(x, W1bf, b1, xp);
    rnn_kernel<<<NBLK_RNN, 256, 0, stream>>>(W2bf, b2, xp, s0, s1, stateF, counter);
    fc3_kernel<<<1, 128, 0, stream>>>(stateF, W3, b3, out);
}
